// FlashCrossAttention_19662360281189
// MI455X (gfx1250) — compile-verified, matrix operand order checked by reading, not run on hardware
//
#include <hip/hip_runtime.h>
#include <hip/hip_bf16.h>

typedef __attribute__((ext_vector_type(16))) _Float16     v16h;
typedef __attribute__((ext_vector_type(8)))  _Float16     v8h;
typedef __attribute__((ext_vector_type(8)))  float        v8f;
typedef __attribute__((ext_vector_type(4)))  unsigned int u32x4;
typedef __attribute__((ext_vector_type(8)))  int          i32x8;
typedef __attribute__((ext_vector_type(4)))  int          i32x4;

#define B_   4
#define TD   2048
#define TE   2048
#define DM   1024
#define NH   16
#define HD   64

#if __has_builtin(__builtin_amdgcn_tensor_load_to_lds) && \
    __has_builtin(__builtin_amdgcn_s_wait_tensorcnt)
#define CDNA5_HAS_TDM 1
#else
#define CDNA5_HAS_TDM 0
#endif

// ---------------------------------------------------------------------------
// helpers
// ---------------------------------------------------------------------------
static __device__ __forceinline__ v16h ld_frag16(const _Float16* p0, const _Float16* p1) {
  // CDNA5 16-bit A/B fragment: 16 halves per lane as two contiguous 8-half runs
  v8h a = *(const v8h*)p0;
  v8h b = *(const v8h*)p1;
  v16h r;
#pragma unroll
  for (int i = 0; i < 8; ++i) { r[i] = a[i]; r[i + 8] = b[i]; }
  return r;
}

static __device__ __forceinline__ v8f wmma32(v16h a, v16h b, v8f c) {
  return __builtin_amdgcn_wmma_f32_16x16x32_f16(false, a, false, b, (short)0, c,
                                                false, false);
}

#if CDNA5_HAS_TDM
// TDM: DMA a 64x64 f16 tile (row stride 64 elems in global) into LDS,
// padding every 32 DWORDs (one 64-half row) with 4 DWORDs -> LDS stride 72 halves.
static __device__ __forceinline__ void tdm_load_tile_64x64_pad72(
    const _Float16* gsrc, void* lds_dst) {
  const unsigned long long ga = (unsigned long long)(uintptr_t)gsrc;
  const unsigned lds = (unsigned)(uintptr_t)lds_dst;
  u32x4 g0 = {};
  g0[0] = 1u;                                           // count=1 (valid user D#)
  g0[1] = lds;                                          // lds_addr (bytes)
  g0[2] = (unsigned)(ga & 0xFFFFFFFFu);                 // global_addr[31:0]
  g0[3] = (unsigned)((ga >> 32) & 0x1FFFFFFu)           // global_addr[56:32]
        | (2u << 30);                                   // type=2 ("image")
  i32x8 g1 = {};
  g1[0] = (1 << 16)                                     // data_size = 2 bytes
        | (1 << 20)                                     // pad_enable
        | (4 << 22)                                     // pad_interval: 32 DWORDs
        | (3 << 25);                                    // pad_amount : 4 DWORDs
  g1[1] = (64 << 16);                                   // tensor_dim0 = 64 (low16)
  g1[2] = 0;                                            // tensor_dim0 hi / dim1 lo
  g1[3] = 64 | (64 << 16);                              // tensor_dim1 hi=64<<16 blocks, tile_dim0=64
  g1[4] = 64;                                           // tile_dim1 = 64
  g1[5] = 64;                                           // tensor_dim0_stride = 64
  g1[6] = 0;
  g1[7] = 0;
  i32x4 g2 = {};
  i32x4 g3 = {};
#if __clang_major__ >= 23
  i32x8 g4 = {};
  __builtin_amdgcn_tensor_load_to_lds(g0, g1, g2, g3, g4, 0);
#else
  __builtin_amdgcn_tensor_load_to_lds(g0, g1, g2, g3, 0);
#endif
}
#endif

// ---------------------------------------------------------------------------
// RMSNorm: x (f32, row of 1024) -> xn (f16)
// ---------------------------------------------------------------------------
__global__ __launch_bounds__(256) void rmsnorm_f16_kernel(
    const float* __restrict__ x, const float* __restrict__ w,
    _Float16* __restrict__ xn) {
  __shared__ float red[256];
  const int row = blockIdx.x;
  const float* xr = x + (size_t)row * DM;
  float s = 0.f;
  for (int j = threadIdx.x; j < DM; j += 256) { float v = xr[j]; s += v * v; }
  red[threadIdx.x] = s;
  __syncthreads();
  for (int off = 128; off > 0; off >>= 1) {
    if (threadIdx.x < off) red[threadIdx.x] += red[threadIdx.x + off];
    __syncthreads();
  }
  const float rms = rsqrtf(red[0] * (1.0f / DM) + 1e-5f);
  _Float16* o = xn + (size_t)row * DM;
  for (int j = threadIdx.x; j < DM; j += 256)
    o[j] = (_Float16)(xr[j] * rms * w[j]);
}

// ---------------------------------------------------------------------------
// generic f32 -> f16 convert
// ---------------------------------------------------------------------------
__global__ __launch_bounds__(256) void f32_to_f16_kernel(
    const float* __restrict__ in, _Float16* __restrict__ out, int n) {
  int i = blockIdx.x * 256 + threadIdx.x;
  if (i < n) out[i] = (_Float16)in[i];
}

// ---------------------------------------------------------------------------
// GEMM: C[M,N] = A[M,K] @ W[N,K]^T   (A,W f16; acc f32)
// block tile 128x64, 256 threads = 8 waves; each wave: 16 rows x 64 cols
// (4 accumulators, A-fragment reused across 4 B-fragments -> 4 wmma / 10 ds).
// EPI==0: store f16 row-major.  EPI==1: store f32 + residual.
// ---------------------------------------------------------------------------
template <int EPI>
__global__ __launch_bounds__(256) void gemm_rt_kernel(
    const _Float16* __restrict__ A, const _Float16* __restrict__ W,
    void* __restrict__ Cout, const float* __restrict__ res,
    int M, int N, int K) {
  __shared__ _Float16 As[128 * 40];  // 128 rows x 32 k, stride 40 (16B aligned)
  __shared__ _Float16 Ws[64 * 40];
  const int tid  = threadIdx.x;
  const int wid  = tid >> 5;          // 0..7 -> m sub-tile (16 rows each)
  const int lane = tid & 31;
  const int l16  = lane & 15;
  const int hi   = (lane >> 4) & 1;
  const int m0   = blockIdx.y * 128;
  const int n0   = blockIdx.x * 64;
  const int d0   = hi * 8;
  const int c8   = (tid & 3) * 8;     // 0,8,16,24

  v8f acc[4];
#pragma unroll
  for (int j = 0; j < 4; ++j) acc[j] = (v8f){};

  for (int k0 = 0; k0 < K; k0 += 32) {
    // stage A (128x32) and W (64x32)
#pragma unroll
    for (int s = 0; s < 2; ++s) {
      const int row = (tid + s * 256) >> 2;   // 0..127
      *(v8h*)&As[row * 40 + c8] =
          *(const v8h*)&A[(size_t)(m0 + row) * K + k0 + c8];
    }
    {
      const int row = tid >> 2;               // 0..63
      *(v8h*)&Ws[row * 40 + c8] =
          *(const v8h*)&W[(size_t)(n0 + row) * K + k0 + c8];
      if (k0 + 32 < K) {
        __builtin_prefetch(&A[(size_t)(m0 + row) * K + k0 + 32 + c8], 0, 0);
        __builtin_prefetch(&W[(size_t)(n0 + row) * K + k0 + 32 + c8], 0, 0);
      }
    }
    __syncthreads();
    v16h af = ld_frag16(&As[(wid * 16 + l16) * 40 + d0],
                        &As[(wid * 16 + l16) * 40 + d0 + 16]);
#pragma unroll
    for (int j = 0; j < 4; ++j) {
      v16h bf = ld_frag16(&Ws[(j * 16 + l16) * 40 + d0],
                          &Ws[(j * 16 + l16) * 40 + d0 + 16]);
      acc[j] = wmma32(af, bf, acc[j]);
    }
    __syncthreads();
  }

#pragma unroll
  for (int r = 0; r < 8; ++r) {
    const int mg = m0 + wid * 16 + r + hi * 8;  // C layout: M = vgpr + 8*(lane>=16)
#pragma unroll
    for (int j = 0; j < 4; ++j) {
      const int ng = n0 + j * 16 + l16;
      if (EPI == 0) {
        ((_Float16*)Cout)[(size_t)mg * N + ng] = (_Float16)acc[j][r];
      } else {
        ((float*)Cout)[(size_t)mg * N + ng] =
            acc[j][r] + res[(size_t)mg * N + ng];
      }
    }
  }
}

// ---------------------------------------------------------------------------
// RoPE on Q: q_raw [B*TD, DM] (f16) -> qh [B,H,TD,HD] (f16)
// ---------------------------------------------------------------------------
static __device__ __forceinline__ float rope_apply(float v, float vp, float sgn,
                                                   int t, int fi) {
  const float inv = __powf(10000.0f, -(float)fi * (1.0f / 32.0f));
  const float ang = (float)t * inv;
  return v * __cosf(ang) + sgn * vp * __sinf(ang);
}

__global__ __launch_bounds__(256) void rope_q_kernel(
    const _Float16* __restrict__ qr, _Float16* __restrict__ qh) {
  const int i = blockIdx.x * 256 + threadIdx.x;  // ((b*NH+h)*TD+t)*HD+d
  const int d = i & 63;
  const int t = (i >> 6) & (TD - 1);
  const int h = (i >> 17) & (NH - 1);
  const int b = i >> 21;
  const size_t row = (size_t)(b * TD + t) * DM + h * 64;
  const int dp = (d < 32) ? d + 32 : d - 32;
  const float sgn = (d < 32) ? -1.0f : 1.0f;
  const float v  = (float)qr[row + d];
  const float vp = (float)qr[row + dp];
  qh[i] = (_Float16)rope_apply(v, vp, sgn, t, d & 31);
}

// kv_raw [B*TE, 2*DM]: k = cols [0,1024), v = cols [1024,2048)
__global__ __launch_bounds__(256) void rope_kv_kernel(
    const _Float16* __restrict__ kvr, _Float16* __restrict__ kh,
    _Float16* __restrict__ vh) {
  const int i = blockIdx.x * 256 + threadIdx.x;  // ((b*NH+h)*TE+t)*HD+d
  const int d = i & 63;
  const int t = (i >> 6) & (TE - 1);
  const int h = (i >> 17) & (NH - 1);
  const int b = i >> 21;
  const size_t row = (size_t)(b * TE + t) * (2 * DM);
  const int dp = (d < 32) ? d + 32 : d - 32;
  const float sgn = (d < 32) ? -1.0f : 1.0f;
  const float kv  = (float)kvr[row + h * 64 + d];
  const float kvp = (float)kvr[row + h * 64 + dp];
  kh[i] = (_Float16)rope_apply(kv, kvp, sgn, t, d & 31);
  vh[i] = kvr[row + DM + h * 64 + d];
}

// ---------------------------------------------------------------------------
// Flash cross-attention: per block = (b,h, 64 q rows), 4 waves x 16 rows.
// K tile staged by the Tensor Data Mover (wave 0, TENSORcnt); V^T by VALU.
// qh/kh/vh: [B,H,T,64] f16.  aout: [B*TD, DM] f16 (heads re-interleaved).
// ---------------------------------------------------------------------------
__global__ __launch_bounds__(128) void flash_attn_kernel(
    const _Float16* __restrict__ qh, const _Float16* __restrict__ kh,
    const _Float16* __restrict__ vh, _Float16* __restrict__ aout) {
  __shared__ _Float16 Ks[64 * 72];       // 64 keys x 64 d (stride 72)
  __shared__ _Float16 Vt[64 * 72];       // 64 d   x 64 keys (stride 72)
  __shared__ _Float16 Ps[4 * 16 * 72];   // per-wave P tile: 16 rows x 64 keys
  const int tid  = threadIdx.x;
  const int wid  = tid >> 5;
  const int lane = tid & 31;
  const int l16  = lane & 15;
  const int hi   = (lane >> 4) & 1;
  const int bh   = blockIdx.y;           // b*NH + h
  const int b    = bh >> 4;
  const int h    = bh & 15;
  const int t0   = blockIdx.x * 64;
  const int d0   = hi * 8;
  const float scale = 0.125f;            // 1/sqrt(64)

  // Q fragments live in registers for the whole kernel (K = 64 = 2 wmma steps)
  const size_t qbase = ((size_t)bh * TD + t0 + wid * 16) * HD;
  const _Float16* qp = qh + qbase + (size_t)l16 * HD;
  v16h qf0 = ld_frag16(qp + d0,      qp + d0 + 16);
  v16h qf1 = ld_frag16(qp + 32 + d0, qp + 32 + d0 + 16);

  float mrow[8], lrow[8];
  v8f oacc[4];
#pragma unroll
  for (int r = 0; r < 8; ++r) { mrow[r] = -1e30f; lrow[r] = 0.f; }
#pragma unroll
  for (int j = 0; j < 4; ++j) oacc[j] = (v8f){};

  _Float16* Pw = &Ps[wid * 16 * 72];

  for (int kb = 0; kb < TE; kb += 64) {
    // ---- stage K tile (TDM) and V^T tile (VALU) into LDS ----
#if CDNA5_HAS_TDM
    if (wid == 0)
      tdm_load_tile_64x64_pad72(&kh[((size_t)bh * TE + kb) * HD], (void*)&Ks[0]);
#endif
    for (int s = tid; s < 64 * 8; s += 128) {
      const int row = s >> 3;
      const int c8  = (s & 7) * 8;
      const size_t g = ((size_t)bh * TE + kb + row) * HD + c8;
#if !CDNA5_HAS_TDM
      v8h k8 = *(const v8h*)&kh[g];
      *(v8h*)&Ks[row * 72 + c8] = k8;
#endif
      v8h v8v = *(const v8h*)&vh[g];
#pragma unroll
      for (int e = 0; e < 8; ++e) Vt[(c8 + e) * 72 + row] = v8v[e];
    }
#if CDNA5_HAS_TDM
    if (wid == 0) __builtin_amdgcn_s_wait_tensorcnt(0);
#endif
    __syncthreads();

    // ---- scores S[16 x 64] = Q @ K^T ----
    v8f sacc[4];
#pragma unroll
    for (int j = 0; j < 4; ++j) {
      v8f a = {};
      v16h bf0 = ld_frag16(&Ks[(j * 16 + l16) * 72 + d0],
                           &Ks[(j * 16 + l16) * 72 + d0 + 16]);
      a = wmma32(qf0, bf0, a);
      v16h bf1 = ld_frag16(&Ks[(j * 16 + l16) * 72 + 32 + d0],
                           &Ks[(j * 16 + l16) * 72 + 32 + d0 + 16]);
      a = wmma32(qf1, bf1, a);
      sacc[j] = a;
    }
#pragma unroll
    for (int j = 0; j < 4; ++j)
#pragma unroll
      for (int r = 0; r < 8; ++r) sacc[j][r] *= scale;

    // ---- online softmax: row reductions across the 16-lane half-groups ----
    float alpha[8];
#pragma unroll
    for (int r = 0; r < 8; ++r) {
      float cm = fmaxf(fmaxf(sacc[0][r], sacc[1][r]),
                       fmaxf(sacc[2][r], sacc[3][r]));
      cm = fmaxf(cm, __shfl_xor(cm, 1));
      cm = fmaxf(cm, __shfl_xor(cm, 2));
      cm = fmaxf(cm, __shfl_xor(cm, 4));
      cm = fmaxf(cm, __shfl_xor(cm, 8));
      const float mnew = fmaxf(mrow[r], cm);
      alpha[r] = __expf(mrow[r] - mnew);
      mrow[r] = mnew;
    }
#pragma unroll
    for (int r = 0; r < 8; ++r) {
      float rs = 0.f;
#pragma unroll
      for (int j = 0; j < 4; ++j) {
        const float p = __expf(sacc[j][r] - mrow[r]);
        rs += p;
        // C layout -> LDS: row = r + 8*hi, col = j*16 + l16 (wave-private, DS in-order)
        Pw[(r + hi * 8) * 72 + j * 16 + l16] = (_Float16)p;
      }
      rs += __shfl_xor(rs, 1);
      rs += __shfl_xor(rs, 2);
      rs += __shfl_xor(rs, 4);
      rs += __shfl_xor(rs, 8);
      lrow[r] = lrow[r] * alpha[r] + rs;
#pragma unroll
      for (int j = 0; j < 4; ++j) oacc[j][r] *= alpha[r];
    }

    // ---- O += P @ V  (contraction over 64 keys = 2 wmma k-steps) ----
#pragma unroll
    for (int s2 = 0; s2 < 2; ++s2) {
      const int kk = s2 * 32 + d0;
      v16h pf = ld_frag16(&Pw[l16 * 72 + kk], &Pw[l16 * 72 + kk + 16]);
#pragma unroll
      for (int j = 0; j < 4; ++j) {
        v16h vf = ld_frag16(&Vt[(j * 16 + l16) * 72 + kk],
                            &Vt[(j * 16 + l16) * 72 + kk + 16]);
        oacc[j] = wmma32(pf, vf, oacc[j]);
      }
    }
    __syncthreads();
  }

  // ---- epilogue: normalize, re-interleave heads into [B*TD, DM] f16 ----
#pragma unroll
  for (int r = 0; r < 8; ++r) {
    const int tg = t0 + wid * 16 + r + hi * 8;
    const float inv = 1.f / lrow[r];
    const size_t rowoff = ((size_t)b * TD + tg) * DM + h * HD + l16;
#pragma unroll
    for (int j = 0; j < 4; ++j)
      aout[rowoff + j * 16] = (_Float16)(oacc[j][r] * inv);
  }
}

// ---------------------------------------------------------------------------
// launch
// ---------------------------------------------------------------------------
extern "C" void kernel_launch(void* const* d_in, const int* in_sizes, int n_in,
                              void* d_out, int out_size, void* d_ws, size_t ws_size,
                              hipStream_t stream) {
  (void)in_sizes; (void)n_in; (void)out_size; (void)ws_size;
  const float* x   = (const float*)d_in[0];  // [B,TD,DM]
  const float* enc = (const float*)d_in[1];  // [B,TE,DM]
  const float* nw  = (const float*)d_in[2];  // [DM]
  const float* Wq  = (const float*)d_in[3];  // [DM,DM]
  const float* Wkv = (const float*)d_in[4];  // [2DM,DM]
  const float* Wo  = (const float*)d_in[5];  // [DM,DM]

  char* ws = (char*)d_ws;
  size_t off = 0;
  auto take = [&](size_t bytes) {
    char* p = ws + off;
    off += (bytes + 255) & ~(size_t)255;
    return p;
  };
  const size_t MQ = (size_t)B_ * TD;        // 8192 rows
  _Float16* xn    = (_Float16*)take(MQ * DM * 2);
  _Float16* ench  = (_Float16*)take(MQ * DM * 2);
  _Float16* Wqh   = (_Float16*)take((size_t)DM * DM * 2);
  _Float16* Wkvh  = (_Float16*)take((size_t)2 * DM * DM * 2);
  _Float16* Woh   = (_Float16*)take((size_t)DM * DM * 2);
  _Float16* qraw  = (_Float16*)take(MQ * DM * 2);
  _Float16* kvraw = (_Float16*)take(MQ * 2 * DM * 2);
  _Float16* qhp   = (_Float16*)take(MQ * DM * 2);
  _Float16* khp   = (_Float16*)take(MQ * DM * 2);
  _Float16* vhp   = (_Float16*)take(MQ * DM * 2);
  _Float16* aoutp = (_Float16*)take(MQ * DM * 2);

  // 1) norm + precision conversion
  rmsnorm_f16_kernel<<<(int)MQ, 256, 0, stream>>>(x, nw, xn);
  {
    const int ne = B_ * TE * DM;
    f32_to_f16_kernel<<<(ne + 255) / 256, 256, 0, stream>>>(enc, ench, ne);
    f32_to_f16_kernel<<<(DM * DM + 255) / 256, 256, 0, stream>>>(Wq, Wqh, DM * DM);
    f32_to_f16_kernel<<<(2 * DM * DM + 255) / 256, 256, 0, stream>>>(Wkv, Wkvh, 2 * DM * DM);
    f32_to_f16_kernel<<<(DM * DM + 255) / 256, 256, 0, stream>>>(Wo, Woh, DM * DM);
  }

  // 2) projections (WMMA GEMMs, 128x64 block tiles)
  gemm_rt_kernel<0><<<dim3(DM / 64, (int)(MQ / 128)), 256, 0, stream>>>(
      xn, Wqh, qraw, nullptr, (int)MQ, DM, DM);
  gemm_rt_kernel<0><<<dim3(2 * DM / 64, (int)(MQ / 128)), 256, 0, stream>>>(
      ench, Wkvh, kvraw, nullptr, (int)MQ, 2 * DM, DM);

  // 3) RoPE + head-major relayout
  const int nrope = B_ * NH * TD * HD;
  rope_q_kernel<<<nrope / 256, 256, 0, stream>>>(qraw, qhp);
  rope_kv_kernel<<<nrope / 256, 256, 0, stream>>>(kvraw, khp, vhp);

  // 4) flash attention (WMMA + TDM K staging, online softmax)
  flash_attn_kernel<<<dim3(TD / 64, B_ * NH), 128, 0, stream>>>(qhp, khp, vhp, aoutp);

  // 5) output projection + residual (WMMA GEMM, f32 out)
  gemm_rt_kernel<1><<<dim3(DM / 64, (int)(MQ / 128)), 256, 0, stream>>>(
      aoutp, Woh, d_out, x, (int)MQ, DM, DM);
}
